// ArcFC_50869592654076
// MI455X (gfx1250) — matrix operand order repaired and checked by reading, not compiled, not run on hardware
//
#include <hip/hip_runtime.h>
#include <math.h>

#define Bdim 512
#define Ddim 512
#define Cdim 100000
#define BN 64
#define LDS_STRIDE 520   // 512 + 8 bf16 pad -> 260 dwords/row, stride%64banks = 4

typedef __bf16 bf16_t;
typedef __bf16 bf16x8 __attribute__((ext_vector_type(8)));
typedef __bf16 v16bf  __attribute__((ext_vector_type(16)));
typedef float  v8f    __attribute__((ext_vector_type(8)));

#define COS_M_C 0.8775825618903728f
#define SIN_M_C 0.479425538604203f
#define TH_C   (-0.8775825618903728f)
#define MM_C    0.2397127693021015f
#define SSCALE  30.0f

// ---- fragment loaders (layouts per CDNA5 ISA §7.12.2) ----
// 16-bit A 16x32: lanes 0-15 hold K {k0..k0+7, k0+16..k0+23}, lanes 16-31 hold
// K {k0+8..k0+15, k0+24..k0+31} for row M = lane%16.
__device__ inline v16bf load_a_frag(const bf16_t* __restrict__ row, int k0, int half) {
  union { v16bf v; bf16x8 h[2]; } u;
  u.h[0] = *(const bf16x8*)(row + k0 + half * 8);
  u.h[1] = *(const bf16x8*)(row + k0 + 16 + half * 8);
  return u.v;
}
// 16-bit B 32x16: lane holds column N = lane%16, K {k0+half*8..+7, k0+16+half*8..+7}
// (same K order as A).
__device__ inline v16bf load_b_frag(const bf16_t* __restrict__ row, int k0, int half) {
  union { v16bf v; bf16x8 h[2]; } u;
  u.h[0] = *(const bf16x8*)(row + k0 + half * 8);
  u.h[1] = *(const bf16x8*)(row + k0 + 16 + half * 8);
  return u.v;
}

// ---- kernel 1: normalize x rows, split into bf16 hi/lo into workspace ----
__global__ __launch_bounds__(256) void xprep_kernel(const float* __restrict__ x,
                                                    bf16_t* __restrict__ xh,
                                                    bf16_t* __restrict__ xl) {
  __shared__ float red[256];
  const int row = blockIdx.x;
  const int t = threadIdx.x;
  const float* xr = x + (size_t)row * Ddim;
  float v0 = xr[t];
  float v1 = xr[t + 256];
  red[t] = v0 * v0 + v1 * v1;
  __syncthreads();
  for (int s = 128; s > 0; s >>= 1) {
    if (t < s) red[t] += red[t + s];
    __syncthreads();
  }
  const float inv = 1.0f / fmaxf(sqrtf(red[0]), 1e-12f);
  float a = v0 * inv, b = v1 * inv;
  bf16_t ah = (bf16_t)a, bh = (bf16_t)b;
  xh[(size_t)row * Ddim + t] = ah;
  xh[(size_t)row * Ddim + t + 256] = bh;
  xl[(size_t)row * Ddim + t] = (bf16_t)(a - (float)ah);
  xl[(size_t)row * Ddim + t + 256] = (bf16_t)(b - (float)bh);
}

// ---- kernel 2: fused normalize-W + split-bf16 WMMA GEMM + ArcFace epilogue ----
__global__ __launch_bounds__(256) void arcface_gemm_kernel(
    const float* __restrict__ W, const bf16_t* __restrict__ xh,
    const bf16_t* __restrict__ xl, const int* __restrict__ label,
    float* __restrict__ out) {
  __shared__ bf16_t sWh[BN * LDS_STRIDE];
  __shared__ bf16_t sWl[BN * LDS_STRIDE];
  __shared__ int sLab[Bdim];

  const int t = threadIdx.x;
  const int cb = blockIdx.x * BN;

  // stage labels for the whole batch
  sLab[t] = label[t];
  sLab[t + 256] = label[t + 256];

  // --- phase 1: load BN rows of W, L2-normalize, split to bf16 hi/lo in LDS ---
  {
    const int r = t >> 2;     // 0..63 : row within block
    const int seg = t & 3;    // 0..3  : 128-col segment
    const int gr = cb + r;
    float ss = 0.0f;
    const float4* p = (const float4*)(W + (size_t)gr * Ddim + seg * 128);
    if (gr < Cdim) {
#pragma unroll
      for (int i = 0; i < 32; ++i) {
        float4 v = p[i];
        ss += v.x * v.x + v.y * v.y + v.z * v.z + v.w * v.w;
      }
    }
    // reduce across the 4 lanes covering one row (lanes 4r..4r+3, same wave)
    ss += __shfl_xor(ss, 1, 32);
    ss += __shfl_xor(ss, 2, 32);
    const float inv = 1.0f / fmaxf(sqrtf(ss), 1e-12f);
    bf16_t* dh = sWh + r * LDS_STRIDE + seg * 128;
    bf16_t* dl = sWl + r * LDS_STRIDE + seg * 128;
    if (gr < Cdim) {
#pragma unroll 4
      for (int i = 0; i < 32; ++i) {
        float4 v = p[i];
        float f0 = v.x * inv, f1 = v.y * inv, f2 = v.z * inv, f3 = v.w * inv;
        bf16_t h0 = (bf16_t)f0, h1 = (bf16_t)f1, h2 = (bf16_t)f2, h3 = (bf16_t)f3;
        dh[4 * i + 0] = h0; dh[4 * i + 1] = h1;
        dh[4 * i + 2] = h2; dh[4 * i + 3] = h3;
        dl[4 * i + 0] = (bf16_t)(f0 - (float)h0);
        dl[4 * i + 1] = (bf16_t)(f1 - (float)h1);
        dl[4 * i + 2] = (bf16_t)(f2 - (float)h2);
        dl[4 * i + 3] = (bf16_t)(f3 - (float)h3);
      }
    } else {
#pragma unroll 8
      for (int i = 0; i < 128; ++i) { dh[i] = (bf16_t)0.0f; dl[i] = (bf16_t)0.0f; }
    }
  }
  __syncthreads();

  // --- phase 2: WMMA main loop ---
  const int lane = t & 31;
  const int wave = t >> 5;      // 0..7 -> 64 batch rows each
  const int half = lane >> 4;
  const int lrow = lane & 15;

  v8f acc[4][4];
  const v8f vzero = {0.f, 0.f, 0.f, 0.f, 0.f, 0.f, 0.f, 0.f};
#pragma unroll
  for (int i = 0; i < 4; ++i)
#pragma unroll
    for (int j = 0; j < 4; ++j) acc[i][j] = vzero;

  const bf16_t* xhr[4];
  const bf16_t* xlr[4];
#pragma unroll
  for (int mt = 0; mt < 4; ++mt) {
    const int M = wave * 64 + mt * 16 + lrow;
    xhr[mt] = xh + (size_t)M * Ddim;
    xlr[mt] = xl + (size_t)M * Ddim;
  }
  const bf16_t* swh = sWh + lrow * LDS_STRIDE;
  const bf16_t* swl = sWl + lrow * LDS_STRIDE;

  for (int k0 = 0; k0 < Ddim; k0 += 32) {
    v16bf bh[4], bl[4];
#pragma unroll
    for (int nt = 0; nt < 4; ++nt) {
      bh[nt] = load_b_frag(swh + nt * 16 * LDS_STRIDE, k0, half);
      bl[nt] = load_b_frag(swl + nt * 16 * LDS_STRIDE, k0, half);
    }
#pragma unroll
    for (int mt = 0; mt < 4; ++mt) {
      v16bf ah = load_a_frag(xhr[mt], k0, half);
      v16bf al = load_a_frag(xlr[mt], k0, half);
#pragma unroll
      for (int nt = 0; nt < 4; ++nt) {
        acc[mt][nt] = __builtin_amdgcn_wmma_f32_16x16x32_bf16(
            false, al, false, bh[nt], (short)0, acc[mt][nt], false, false);
        acc[mt][nt] = __builtin_amdgcn_wmma_f32_16x16x32_bf16(
            false, ah, false, bl[nt], (short)0, acc[mt][nt], false, false);
        acc[mt][nt] = __builtin_amdgcn_wmma_f32_16x16x32_bf16(
            false, ah, false, bh[nt], (short)0, acc[mt][nt], false, false);
      }
    }
  }

  // --- phase 3: ArcFace epilogue straight from accumulator layout ---
  // C/D layout: VGPR j, lanes 0-15 -> (M=j, N=lane); lanes 16-31 -> (M=j+8, N=lane-16)
  const int rbase = half ? 8 : 0;
#pragma unroll
  for (int mt = 0; mt < 4; ++mt) {
#pragma unroll
    for (int nt = 0; nt < 4; ++nt) {
      const int Ng = cb + nt * 16 + lrow;
      const bool nvalid = Ng < Cdim;
#pragma unroll
      for (int j = 0; j < 8; ++j) {
        const int M = wave * 64 + mt * 16 + rbase + j;
        float c = acc[mt][nt][j];
        float s = sqrtf(fmaxf(1.0f - c * c, 0.0f));
        float phi = c * COS_M_C - s * SIN_M_C;
        phi = (c > TH_C) ? phi : (c - MM_C);
        float o = (sLab[M] == Ng) ? phi : c;
        if (nvalid) out[(size_t)M * Cdim + Ng] = o * SSCALE;
      }
    }
  }
}

extern "C" void kernel_launch(void* const* d_in, const int* in_sizes, int n_in,
                              void* d_out, int out_size, void* d_ws, size_t ws_size,
                              hipStream_t stream) {
  (void)in_sizes; (void)n_in; (void)out_size; (void)ws_size;
  const float* x = (const float*)d_in[0];
  const float* W = (const float*)d_in[1];
  const int* label = (const int*)d_in[2];
  float* out = (float*)d_out;

  bf16_t* xh = (bf16_t*)d_ws;                       // 512*512 bf16
  bf16_t* xl = xh + (size_t)Bdim * Ddim;            // 512*512 bf16 (1 MB total)

  xprep_kernel<<<Bdim, 256, 0, stream>>>(x, xh, xl);

  const int grid = (Cdim + BN - 1) / BN;            // 1563 workgroups
  arcface_gemm_kernel<<<grid, 256, 0, stream>>>(W, xh, xl, label, out);
}
